// CondensedGraphDecoder_11553462026275
// MI455X (gfx1250) — matrix operand order repaired and checked by reading, not compiled, not run on hardware
//
#include <hip/hip_runtime.h>

// ---------------- problem constants ----------------
#define N_NODES   2048
#define E_EDGES   131072
#define PDIM      512
#define NDIM      256
#define EDIM      128
#define H1DIM     64
#define K_SEL     13107          // int(0.1 * 131072)
#define DEG_THRESH 2048.0f       // 10.0 * 0.1 * 2048

// ---------------- WMMA types (gfx1250, wave32) ----------------
typedef __attribute__((ext_vector_type(16))) __bf16 v16bf;
typedef __attribute__((ext_vector_type(8)))  float  v8f;
typedef __attribute__((ext_vector_type(4)))  int    v4i_;

struct alignas(16) U128 { unsigned x, y, z, w; };
union ABFrag { v16bf v; unsigned u[8]; U128 q[2]; };

// Optional gfx1250 async global->LDS path (guarded; falls back to plain LDS stores)
#if __has_builtin(__builtin_amdgcn_global_load_async_to_lds_b128) && \
    __has_builtin(__builtin_amdgcn_s_wait_asynccnt)
#define USE_ASYNC_LDS 1
typedef __attribute__((address_space(1))) v4i_ gas_v4i;
typedef __attribute__((address_space(3))) v4i_ las_v4i;
#else
#define USE_ASYNC_LDS 0
#endif

static __device__ inline void stage16(void* l, const void* g) {
#if USE_ASYNC_LDS
  __builtin_amdgcn_global_load_async_to_lds_b128((gas_v4i*)g, (las_v4i*)l, 0, 0);
#else
  *(U128*)l = *(const U128*)g;
#endif
}
static __device__ inline void stage_wait() {
#if USE_ASYNC_LDS
  __builtin_amdgcn_s_wait_asynccnt(0);
#endif
}

static __device__ inline unsigned short f2bf(float f) {
  unsigned u = __float_as_uint(f);
  unsigned r = u + 0x7fffu + ((u >> 16) & 1u);   // RNE
  return (unsigned short)(r >> 16);
}

// A fragment: 16x32 bf16 tile, row-major source (contiguous k = 16*kh + s order, B packed to match).
// lane: m = lane&15, kh = lane>>4. Per-lane data is two contiguous 16B runs.
static __device__ inline v16bf load_a_frag(const unsigned short* A, int lda,
                                           int row0, int k0, int lane) {
  int m = lane & 15, kh = (lane >> 4) & 1;
  const unsigned short* base = A + (size_t)(row0 + m) * lda + k0 + kh * 16;
  ABFrag r;
  r.q[0] = *(const U128*)(base);        // K = kh*16 .. kh*16+7
  r.q[1] = *(const U128*)(base + 8);    // K = kh*16+8 .. kh*16+15
  return r.v;
}

// Packed-B fragment load (weights pre-packed into fragment layout).
static __device__ inline v16bf load_b_packed(const U128* Bpk, int ntiles,
                                             int kt, int nt, int lane) {
  const U128* p = Bpk + ((size_t)(kt * ntiles + nt) * 32 + lane) * 2;
  ABFrag r;
  r.q[0] = p[0];
  r.q[1] = p[1];
  return r.v;
}

static __device__ inline void store_tile(float* C, int ldc, int row0, int col0,
                                         int lane, v8f c) {
  int n = lane & 15, g = (lane >> 4) & 1;
#pragma unroll
  for (int v = 0; v < 8; ++v)
    C[(size_t)(row0 + v + g * 8) * ldc + col0 + n] = c[v];
}

// ---------------- cast / init / pack kernels ----------------
__global__ void cast_f32_bf16(const float* __restrict__ src,
                              unsigned short* __restrict__ dst, int n) {
  int i = blockIdx.x * blockDim.x + threadIdx.x;
  if (i < n) dst[i] = f2bf(src[i]);
}

__global__ void fill_f32(float* __restrict__ p, float v, int n) {
  int i = blockIdx.x * blockDim.x + threadIdx.x;
  if (i < n) p[i] = v;
}

// Pre-pack row-major bf16 B [K x N] into WMMA B-fragment layout:
// frag (kt, nt): 32 lanes x 8 dwords, contiguous.
__global__ void pack_b(const unsigned short* __restrict__ B, int K, int N,
                       U128* __restrict__ out) {
  int idx = blockIdx.x * blockDim.x + threadIdx.x;
  int ntiles = N >> 4;
  int total = (K >> 5) * ntiles * 32;
  if (idx >= total) return;
  int lane = idx & 31, tile = idx >> 5;
  int nt = tile % ntiles, kt = tile / ntiles;
  int n = lane & 15, koff = ((lane >> 4) & 1) << 4;
  const unsigned short* base = B + (size_t)(kt * 32 + koff) * N + nt * 16 + n;
  ABFrag f;
#pragma unroll
  for (int v = 0; v < 8; ++v) {
    unsigned lo = base[(2 * v) * N];
    unsigned hi = base[(2 * v + 1) * N];
    f.u[v] = lo | (hi << 16);
  }
  U128* o = out + (size_t)idx * 2;
  o[0] = f.q[0];
  o[1] = f.q[1];
}

// per-row: cast prototypes to bf16 and produce L2-normalized bf16 copy
__global__ void proto_cast_norm(const float* __restrict__ P,
                                unsigned short* __restrict__ pb,
                                unsigned short* __restrict__ pnb) {
  __shared__ float red[256];
  int r = blockIdx.x, t = threadIdx.x;
  float x0 = P[(size_t)r * PDIM + t];
  float x1 = P[(size_t)r * PDIM + t + 256];
  red[t] = x0 * x0 + x1 * x1;
  __syncthreads();
  for (int s = 128; s > 0; s >>= 1) {
    if (t < s) red[t] += red[t + s];
    __syncthreads();
  }
  float inv = 1.0f / fmaxf(sqrtf(red[0]), 1e-8f);
  pb[(size_t)r * PDIM + t]        = f2bf(x0);
  pb[(size_t)r * PDIM + t + 256]  = f2bf(x1);
  pnb[(size_t)r * PDIM + t]       = f2bf(x0 * inv);
  pnb[(size_t)r * PDIM + t + 256] = f2bf(x1 * inv);
}

// ---------------- node GEMM: 2x2 register-blocked, packed weights ----------
__global__ void gemm_node(const unsigned short* __restrict__ A,
                          const U128* __restrict__ Bpk,
                          const float* __restrict__ bias,
                          float* __restrict__ C) {
  int lane = threadIdx.x;
  int row0 = blockIdx.x * 32, col0 = blockIdx.y * 32;
  v8f c00 = {}, c01 = {}, c10 = {}, c11 = {};
  for (int k0 = 0, kt = 0; k0 < PDIM; k0 += 32, ++kt) {
    v16bf a0 = load_a_frag(A, PDIM, row0, k0, lane);
    v16bf a1 = load_a_frag(A, PDIM, row0 + 16, k0, lane);
    v16bf b0 = load_b_packed(Bpk, NDIM >> 4, kt, (col0 >> 4), lane);
    v16bf b1 = load_b_packed(Bpk, NDIM >> 4, kt, (col0 >> 4) + 1, lane);
    c00 = __builtin_amdgcn_wmma_f32_16x16x32_bf16(false, a0, false, b0, (short)0, c00, false, false);
    c01 = __builtin_amdgcn_wmma_f32_16x16x32_bf16(false, a0, false, b1, (short)0, c01, false, false);
    c10 = __builtin_amdgcn_wmma_f32_16x16x32_bf16(false, a1, false, b0, (short)0, c10, false, false);
    c11 = __builtin_amdgcn_wmma_f32_16x16x32_bf16(false, a1, false, b1, (short)0, c11, false, false);
  }
  int n = lane & 15, g = (lane >> 4) & 1;
  float bv0 = bias[col0 + n], bv1 = bias[col0 + 16 + n];
#pragma unroll
  for (int v = 0; v < 8; ++v) {
    size_t r0 = (size_t)(row0 + v + g * 8) * NDIM;
    size_t r1 = (size_t)(row0 + 16 + v + g * 8) * NDIM;
    C[r0 + col0 + n]      = c00[v] + bv0;
    C[r0 + col0 + 16 + n] = c01[v] + bv1;
    C[r1 + col0 + n]      = c10[v] + bv0;
    C[r1 + col0 + 16 + n] = c11[v] + bv1;
  }
}

// ---------------- sim GEMM: LDS-staged (async), 8 waves, 128x64 macro-tile --
__global__ void gemm_sim_lds(const unsigned short* __restrict__ PN,
                             float* __restrict__ S) {
  __shared__ alignas(16) unsigned short Ap[128 * 32];  // 8 KB
  __shared__ alignas(16) unsigned short Bp[64 * 32];   // 4 KB
  int t = threadIdx.x;               // 0..255
  int lane = t & 31, w = t >> 5;     // 8 waves
  int wm = w & 3, wn = w >> 2;       // 4x2 wave grid -> 32x32 each
  int row0 = blockIdx.x * 128, col0 = blockIdx.y * 64;
  v8f c00 = {}, c01 = {}, c10 = {}, c11 = {};

  for (int k0 = 0; k0 < PDIM; k0 += 32) {
    __syncthreads();   // previous iteration's LDS reads done
    // stage A panel: 128 rows x 32 cols -> 512 x 16B chunks, 2 per thread
#pragma unroll
    for (int i = 0; i < 2; ++i) {
      int cch = t * 2 + i, r = cch >> 2, x = cch & 3;
      stage16(Ap + r * 32 + x * 8,
              PN + (size_t)(row0 + r) * PDIM + k0 + x * 8);
    }
    // stage B panel (pn rows used as columns): 64 rows x 32 cols -> 1 chunk/thread
    {
      int r = t >> 2, x = t & 3;
      stage16(Bp + r * 32 + x * 8,
              PN + (size_t)(col0 + r) * PDIM + k0 + x * 8);
    }
    stage_wait();      // this wave's async stores to LDS complete
    __syncthreads();   // all waves' stores visible

    // fragments from LDS (ds_load_b128 pairs)
    int m = lane & 15, kh = (lane >> 4) & 1;
    v16bf a0, a1, b0, b1;
    {
      ABFrag f;
      const U128* p = (const U128*)(Ap + (wm * 32 + m) * 32);
      f.q[0] = p[2 * kh]; f.q[1] = p[2 * kh + 1];
      a0 = f.v;
      p = (const U128*)(Ap + (wm * 32 + 16 + m) * 32);
      f.q[0] = p[2 * kh]; f.q[1] = p[2 * kh + 1];
      a1 = f.v;
    }
    {
      int ko = ((lane >> 4) & 1) << 4;   // element offset 0 or 16
      ABFrag f;
      const U128* p = (const U128*)(Bp + (wn * 32 + m) * 32 + ko);
      f.q[0] = p[0]; f.q[1] = p[1];
      b0 = f.v;
      p = (const U128*)(Bp + (wn * 32 + 16 + m) * 32 + ko);
      f.q[0] = p[0]; f.q[1] = p[1];
      b1 = f.v;
    }
    c00 = __builtin_amdgcn_wmma_f32_16x16x32_bf16(false, a0, false, b0, (short)0, c00, false, false);
    c01 = __builtin_amdgcn_wmma_f32_16x16x32_bf16(false, a0, false, b1, (short)0, c01, false, false);
    c10 = __builtin_amdgcn_wmma_f32_16x16x32_bf16(false, a1, false, b0, (short)0, c10, false, false);
    c11 = __builtin_amdgcn_wmma_f32_16x16x32_bf16(false, a1, false, b1, (short)0, c11, false, false);
  }
  int r0 = row0 + wm * 32, cc0 = col0 + wn * 32;
  store_tile(S, N_NODES, r0,      cc0,      lane, c00);
  store_tile(S, N_NODES, r0,      cc0 + 16, lane, c01);
  store_tile(S, N_NODES, r0 + 16, cc0,      lane, c10);
  store_tile(S, N_NODES, r0 + 16, cc0 + 16, lane, c11);
}

// ---------------- fused edge MLP (packed W1 fragments) ----------------
__global__ void edge_mlp(const unsigned short* __restrict__ EF,
                         const U128* __restrict__ W1pk,
                         const float* __restrict__ b1,
                         const float* __restrict__ W2,
                         const float* __restrict__ b2,
                         float* __restrict__ probs) {
  __shared__ float ls[16 * H1DIM];
  int lane = threadIdx.x;
  int e0 = blockIdx.x * 16;
  v8f c0 = {}, c1 = {}, c2 = {}, c3 = {};
#pragma unroll
  for (int kt = 0; kt < EDIM / 32; ++kt) {
    v16bf a = load_a_frag(EF, EDIM, e0, kt * 32, lane);
    v16bf b;
    b = load_b_packed(W1pk, H1DIM >> 4, kt, 0, lane);
    c0 = __builtin_amdgcn_wmma_f32_16x16x32_bf16(false, a, false, b, (short)0, c0, false, false);
    b = load_b_packed(W1pk, H1DIM >> 4, kt, 1, lane);
    c1 = __builtin_amdgcn_wmma_f32_16x16x32_bf16(false, a, false, b, (short)0, c1, false, false);
    b = load_b_packed(W1pk, H1DIM >> 4, kt, 2, lane);
    c2 = __builtin_amdgcn_wmma_f32_16x16x32_bf16(false, a, false, b, (short)0, c2, false, false);
    b = load_b_packed(W1pk, H1DIM >> 4, kt, 3, lane);
    c3 = __builtin_amdgcn_wmma_f32_16x16x32_bf16(false, a, false, b, (short)0, c3, false, false);
  }
  int n = lane & 15, g = (lane >> 4) & 1;
#pragma unroll
  for (int v = 0; v < 8; ++v) {
    int m = v + g * 8;
    ls[m * H1DIM + 0  + n] = fmaxf(c0[v] + b1[0  + n], 0.0f);
    ls[m * H1DIM + 16 + n] = fmaxf(c1[v] + b1[16 + n], 0.0f);
    ls[m * H1DIM + 32 + n] = fmaxf(c2[v] + b1[32 + n], 0.0f);
    ls[m * H1DIM + 48 + n] = fmaxf(c3[v] + b1[48 + n], 0.0f);
  }
  __syncthreads();
  if (lane < 16) {
    float acc = b2[0];
#pragma unroll 8
    for (int k = 0; k < H1DIM; ++k) acc += ls[lane * H1DIM + k] * W2[k];
    probs[e0 + lane] = 1.0f / (1.0f + expf(-acc));
  }
}

// ---------------- softmax over masked sim rows (in place) ----------------
__global__ void softmax_rows(float* __restrict__ S,
                             const int* __restrict__ labels) {
  __shared__ float red[256];
  int r = blockIdx.x, t = threadIdx.x;
  int lr = labels[r];
  float v[8];
  float mx = -__builtin_inff();
#pragma unroll
  for (int k = 0; k < 8; ++k) {
    int cidx = t + 256 * k;
    float x = S[(size_t)r * N_NODES + cidx];
    x = (labels[cidx] == lr) ? x : 0.0f;   // sim * label_mask
    v[k] = x;
    mx = fmaxf(mx, x);
  }
  red[t] = mx; __syncthreads();
  for (int s = 128; s > 0; s >>= 1) { if (t < s) red[t] = fmaxf(red[t], red[t + s]); __syncthreads(); }
  mx = red[0]; __syncthreads();
  float sum = 0.0f;
#pragma unroll
  for (int k = 0; k < 8; ++k) { v[k] = expf(v[k] - mx); sum += v[k]; }
  red[t] = sum; __syncthreads();
  for (int s = 128; s > 0; s >>= 1) { if (t < s) red[t] += red[t + s]; __syncthreads(); }
  float inv = 1.0f / red[0];
#pragma unroll
  for (int k = 0; k < 8; ++k)
    S[(size_t)r * N_NODES + t + 256 * k] = v[k] * inv;
}

// ---------------- score + sort key ----------------
__global__ void score_kernel(const float* __restrict__ probs,
                             const float* __restrict__ attn,
                             const int* __restrict__ ei,
                             const int* __restrict__ labels,
                             float* __restrict__ scores,
                             unsigned long long* __restrict__ keys) {
  int e = blockIdx.x * blockDim.x + threadIdx.x;
  if (e >= E_EDGES) return;
  int s = ei[e], d = ei[E_EDGES + e];
  float sc;
  if ((unsigned)s < (unsigned)N_NODES && (unsigned)d < (unsigned)N_NODES) {
    float lm = (labels[s] == labels[d]) ? 1.0f : 0.0f;
    sc = probs[e] * attn[(size_t)s * N_NODES + d] * (1.0f + 0.5f * lm);
  } else {
    sc = -__builtin_inff();
  }
  scores[e] = sc;
  unsigned u = __float_as_uint(sc);
  unsigned m = (u & 0x80000000u) ? ~u : (u | 0x80000000u); // monotone ascending
  keys[e] = ((unsigned long long)(~m) << 32) | (unsigned)e; // asc key == desc score
}

__global__ void bitonic_step(unsigned long long* __restrict__ keys,
                             unsigned j, unsigned k) {
  unsigned i = blockIdx.x * blockDim.x + threadIdx.x;
  unsigned ixj = i ^ j;
  if (ixj > i) {
    unsigned long long a = keys[i], b = keys[ixj];
    bool asc = ((i & k) == 0);
    if ((a > b) == asc) { keys[i] = b; keys[ixj] = a; }
  }
}

// ---------------- serial greedy selection (O(1) per step) ----------------
__global__ void greedy_scan(const unsigned long long* __restrict__ keys,
                            const float* __restrict__ scores,
                            const int* __restrict__ ei,
                            float* __restrict__ adj,
                            float* __restrict__ deg,
                            float* __restrict__ sel_i,
                            float* __restrict__ sel_j,
                            float* __restrict__ sel_m) {
  bool stopped = false;
  for (int t = 0; t < K_SEL; ++t) {
    int e = (int)(keys[t] & 0xffffffffu);
    int i = ei[e], j = ei[E_EDGES + e];
    float s = scores[e];
    float oi = -1.0f, oj = -1.0f, om = 0.0f;
    if ((unsigned)i < (unsigned)N_NODES && (unsigned)j < (unsigned)N_NODES) {
      bool take = !stopped && (adj[(size_t)i * N_NODES + j] == 0.0f) && (s > 0.0f);
      if (take) {
        adj[(size_t)i * N_NODES + j] = 1.0f;
        adj[(size_t)j * N_NODES + i] = 1.0f;
        deg[i] += 1.0f;
        if (j != i) deg[j] += 1.0f;
        if (deg[i] > DEG_THRESH) stopped = true;
        oi = (float)i; oj = (float)j; om = 1.0f;
      }
    }
    sel_i[t] = oi; sel_j[t] = oj; sel_m[t] = om;
  }
}

// ---------------- launch ----------------
extern "C" void kernel_launch(void* const* d_in, const int* in_sizes, int n_in,
                              void* d_out, int out_size, void* d_ws, size_t ws_size,
                              hipStream_t stream) {
  const float* prototypes    = (const float*)d_in[0];
  const float* edge_features = (const float*)d_in[1];
  const float* W_node        = (const float*)d_in[2];
  const float* b_node        = (const float*)d_in[3];
  const float* W_e1          = (const float*)d_in[4];
  const float* b_e1          = (const float*)d_in[5];
  const float* W_e2          = (const float*)d_in[6];
  const float* b_e2          = (const float*)d_in[7];
  const int*   edge_index    = (const int*)d_in[8];
  const int*   node_labels   = (const int*)d_in[9];

  char* w = (char*)d_ws;
  unsigned short* protos_bf = (unsigned short*)(w + 0);            // 2 MB
  unsigned short* pn_bf     = (unsigned short*)(w + 2097152);      // 2 MB
  unsigned short* Wnode_bf  = (unsigned short*)(w + 4194304);      // 256 KB
  unsigned short* We1_bf    = (unsigned short*)(w + 4456448);      // 16 KB
  unsigned short* ef_bf     = (unsigned short*)(w + 4472832);      // 32 MB
  float*          attn      = (float*)(w + 38027264);              // 16 MB
  float*          probs     = (float*)(w + 54804480);              // 512 KB
  float*          scores    = (float*)(w + 55328768);              // 512 KB
  unsigned long long* keys  = (unsigned long long*)(w + 55853056); // 1 MB
  float*          deg       = (float*)(w + 56901632);              // 8 KB
  U128*           Wnode_pk  = (U128*)(w + 56909824);               // 256 KB
  U128*           We1_pk    = (U128*)(w + 57171968);               // 16 KB

  float* out = (float*)d_out;
  float* adj_out = out;                                 // N*N
  float* nf_out  = out + (size_t)N_NODES * N_NODES;     // N*ND
  float* sel_i   = nf_out + (size_t)N_NODES * NDIM;     // K_SEL
  float* sel_j   = sel_i + K_SEL;
  float* sel_m   = sel_j + K_SEL;

  // 1) casts + normalization
  proto_cast_norm<<<N_NODES, 256, 0, stream>>>(prototypes, protos_bf, pn_bf);
  cast_f32_bf16<<<(PDIM * NDIM) / 256, 256, 0, stream>>>(W_node, Wnode_bf, PDIM * NDIM);
  cast_f32_bf16<<<(EDIM * H1DIM + 255) / 256, 256, 0, stream>>>(W_e1, We1_bf, EDIM * H1DIM);
  cast_f32_bf16<<<(E_EDGES * EDIM) / 256, 256, 0, stream>>>(edge_features, ef_bf, E_EDGES * EDIM);

  // 2) pre-pack weight matrices into WMMA B-fragment layout
  pack_b<<<((PDIM / 32) * (NDIM / 16) * 32 + 255) / 256, 256, 0, stream>>>(
      Wnode_bf, PDIM, NDIM, Wnode_pk);
  pack_b<<<((EDIM / 32) * (H1DIM / 16) * 32 + 255) / 256, 256, 0, stream>>>(
      We1_bf, EDIM, H1DIM, We1_pk);

  // 3) zero-init state written incrementally
  fill_f32<<<(N_NODES * N_NODES) / 256, 256, 0, stream>>>(adj_out, 0.0f, N_NODES * N_NODES);
  fill_f32<<<N_NODES / 256, 256, 0, stream>>>(deg, 0.0f, N_NODES);

  // 4) WMMA GEMMs
  {
    dim3 g(N_NODES / 32, NDIM / 32);
    gemm_node<<<g, 32, 0, stream>>>(protos_bf, Wnode_pk, b_node, nf_out);
  }
  {
    dim3 g(N_NODES / 128, N_NODES / 64);
    gemm_sim_lds<<<g, 256, 0, stream>>>(pn_bf, attn);
  }

  // 5) masked softmax -> attn (in place)
  softmax_rows<<<N_NODES, 256, 0, stream>>>(attn, node_labels);

  // 6) fused edge MLP (WMMA) -> probs
  edge_mlp<<<E_EDGES / 16, 32, 0, stream>>>(ef_bf, We1_pk, b_e1, W_e2, b_e2, probs);

  // 7) scores + sort keys
  score_kernel<<<E_EDGES / 256, 256, 0, stream>>>(probs, attn, edge_index,
                                                  node_labels, scores, keys);

  // 8) bitonic sort (ascending key == descending score, stable by index)
  for (unsigned k = 2; k <= (unsigned)E_EDGES; k <<= 1)
    for (unsigned j = k >> 1; j > 0; j >>= 1)
      bitonic_step<<<E_EDGES / 256, 256, 0, stream>>>(keys, j, k);

  // 9) serial greedy selection
  greedy_scan<<<1, 1, 0, stream>>>(keys, scores, edge_index, adj_out, deg,
                                   sel_i, sel_j, sel_m);
}
